// GCN_352187318590
// MI455X (gfx1250) — compile-verified, matrix operand order checked by reading, not run on hardware
//
#include <hip/hip_runtime.h>
#include <hip/hip_bf16.h>

// ---------------------------------------------------------------------------
// GCN (3x GCNConv + linear + log_softmax) for MI455X / gfx1250.
// L2-resident, atomic-bound workload; GEMMs on v_wmma_f32_16x16x32_bf16.
// Weights pre-packed into WMMA B-fragment layout (one v16bf load per tile).
// One wave computes a 16 x (16*TN) C stripe: A fragment reused across TN wmmas.
// ---------------------------------------------------------------------------

typedef __bf16 bf16_t;
typedef __attribute__((ext_vector_type(16))) __bf16 v16bf;
typedef __attribute__((ext_vector_type(8)))  __bf16 v8bf;
typedef __attribute__((ext_vector_type(8)))  float  v8f;

#define THREADS 256

// ---------------- degree / normalization ----------------

__global__ void k_init_deg(float* __restrict__ deg, int n) {
    int i = blockIdx.x * blockDim.x + threadIdx.x;
    if (i < n) deg[i] = 1.0f;               // self-loop weight = 1
}

__global__ void k_deg_scatter(const int* __restrict__ dst,
                              const float* __restrict__ ew,
                              float* __restrict__ deg, int E) {
    int e = blockIdx.x * blockDim.x + threadIdx.x;
    if (e < E) atomicAdd(&deg[dst[e]], ew[e]);
}

__global__ void k_dinv(const float* __restrict__ deg,
                       float* __restrict__ dinv, int n) {
    int i = blockIdx.x * blockDim.x + threadIdx.x;
    if (i < n) {
        float d = deg[i];
        dinv[i] = (d > 0.0f) ? rsqrtf(d) : 0.0f;
    }
}

// ---------------- f32 -> bf16 convert (optionally fused ReLU) ----------------

__global__ void k_cvt_bf16(const float* __restrict__ in,
                           bf16_t* __restrict__ out,
                           size_t n, int do_relu) {
    size_t i = (size_t)blockIdx.x * blockDim.x + threadIdx.x;
    if (i < n) {
        float v = in[i];
        if (do_relu) v = fmaxf(v, 0.0f);
        out[i] = (bf16_t)v;
    }
}

// ---------------- weight pack: f32 [K x Nc] -> bf16 WMMA B fragments ---------
// Bp layout: [tn][kk][lane][16 bf16]; GEMM loads one contiguous v16bf per
// (kk, tn) per lane. Fragment elements follow ISA 7.12.2 (wave32, 16-bit B):
//   lane l<16  -> col N=l,  K = {k0..k0+7, k0+16..k0+23}
//   lane l>=16 -> col N=l-16, K = {k0+8..k0+15, k0+24..k0+31}

__global__ void k_pack_w(const float* __restrict__ W,
                         bf16_t* __restrict__ Bp, int K, int Nc) {
    int t = blockIdx.x * blockDim.x + threadIdx.x;
    const int KT = K >> 5, TN = Nc >> 4;
    if (t >= TN * KT * 32) return;
    const int lane = t & 31;
    const int kk   = (t >> 5) % KT;
    const int tn   = (t >> 5) / KT;
    const int half = lane >> 4;
    const int l15  = lane & 15;
    bf16_t* __restrict__ dp = Bp + (size_t)t * 16;
#pragma unroll
    for (int i = 0; i < 8; ++i) {
        const int kLo = kk * 32 + half * 8 + i;
        const int kHi = kLo + 16;
        dp[i]     = (bf16_t)W[(size_t)kLo * Nc + tn * 16 + l15];
        dp[i + 8] = (bf16_t)W[(size_t)kHi * Nc + tn * 16 + l15];
    }
}

// ---------------- WMMA GEMM: C[M x 16*TN] = A[M x K](bf16) * Bp -------------
// One wave per 16-row stripe; sweeps all TN column tiles so the A fragment is
// reused TN times per k-step. K, TN compile-time -> fully unrolled (KT*TN wmma).

template <int K, int TN>
__global__ void k_gemm_wmma_bf16(const bf16_t* __restrict__ A,
                                 const bf16_t* __restrict__ Bp,
                                 float* __restrict__ C, int M) {
    constexpr int KT = K / 32;
    constexpr int Nc = TN * 16;
    const int wave = blockIdx.x * (blockDim.x >> 5) + (threadIdx.x >> 5);
    const int lane = threadIdx.x & 31;
    if (wave >= (M >> 4)) return;             // wave-uniform: EXEC stays all-1s

    const int row0 = wave << 4;
    const int half = lane >> 4;
    const int l15  = lane & 15;
    const bf16_t* __restrict__ Arow = A + (size_t)(row0 + l15) * K;

    v8f acc[TN];
#pragma unroll
    for (int tn = 0; tn < TN; ++tn) acc[tn] = (v8f){};

#pragma unroll
    for (int kk = 0; kk < KT; ++kk) {
        const int k0 = kk * 32;
        // A fragment: two contiguous 16B chunks per lane (global_load_b128 x2)
        v8bf alo = *(const v8bf*)(Arow + k0 + half * 8);
        v8bf ahi = *(const v8bf*)(Arow + k0 + 16 + half * 8);
        v16bf a  = __builtin_shufflevector(alo, ahi,
                       0, 1, 2, 3, 4, 5, 6, 7, 8, 9, 10, 11, 12, 13, 14, 15);
#pragma unroll
        for (int tn = 0; tn < TN; ++tn) {
            v16bf b = *(const v16bf*)(Bp + (((size_t)tn * KT + kk) * 32 + lane) * 16);
            acc[tn] = __builtin_amdgcn_wmma_f32_16x16x32_bf16(
                          false, a, false, b, (short)0, acc[tn], false, false);
        }
    }

#pragma unroll
    for (int tn = 0; tn < TN; ++tn) {
#pragma unroll
        for (int r = 0; r < 8; ++r) {
            C[(size_t)(row0 + r + half * 8) * Nc + tn * 16 + l15] = acc[tn][r];
        }
    }
}

// ---------------- aggregation: out = bias + self-loop + edge scatter ---------

__global__ void k_agg_init(const float* __restrict__ h,
                           const float* __restrict__ dinv,
                           const float* __restrict__ bias,
                           float* __restrict__ out, int N, int F) {
    size_t idx = (size_t)blockIdx.x * blockDim.x + threadIdx.x;
    if (idx < (size_t)N * F) {
        int i = (int)(idx / F);
        int f = (int)(idx % F);
        float di = dinv[i];
        out[idx] = bias[f] + di * di * h[idx];    // self-loop: norm = dinv[i]^2
    }
}

// One wave per edge. F==128 fast path: per-lane float4 gather (one coalesced
// 512B row sweep) + 4 f32 atomics into the L2-resident accumulator.
__global__ void k_edge_scatter(const float* __restrict__ h,
                               const int* __restrict__ src,
                               const int* __restrict__ dst,
                               const float* __restrict__ ew,
                               const float* __restrict__ dinv,
                               float* __restrict__ out, int E, int F) {
    int e    = blockIdx.x * (blockDim.x >> 5) + (threadIdx.x >> 5);
    int lane = threadIdx.x & 31;
    if (e >= E) return;
    int s = src[e], d = dst[e];
    float c = dinv[s] * ew[e] * dinv[d];
    const float* __restrict__ hs = h + (size_t)s * F;
    float* __restrict__ od = out + (size_t)d * F;
    if (F == 128) {
        const float4 hv = *(const float4*)(hs + lane * 4);
        atomicAdd(&od[lane * 4 + 0], c * hv.x);
        atomicAdd(&od[lane * 4 + 1], c * hv.y);
        atomicAdd(&od[lane * 4 + 2], c * hv.z);
        atomicAdd(&od[lane * 4 + 3], c * hv.w);
    } else {
        for (int f = lane; f < F; f += 32)
            atomicAdd(&od[f], c * hs[f]);
    }
}

// ---------------- log_softmax over 64 logits, one wave per row ---------------

__global__ void k_logsoftmax64(const float* __restrict__ h,
                               const float* __restrict__ bias,
                               float* __restrict__ out, int N) {
    int row  = blockIdx.x * (blockDim.x >> 5) + (threadIdx.x >> 5);
    int lane = threadIdx.x & 31;
    if (row >= N) return;
    const float* hr = h + (size_t)row * 64;
    float v0 = hr[lane]      + bias[lane];
    float v1 = hr[lane + 32] + bias[lane + 32];
    float m = fmaxf(v0, v1);
#pragma unroll
    for (int off = 16; off > 0; off >>= 1)
        m = fmaxf(m, __shfl_xor(m, off, 32));
    float s = __expf(v0 - m) + __expf(v1 - m);
#pragma unroll
    for (int off = 16; off > 0; off >>= 1)
        s += __shfl_xor(s, off, 32);
    float ls = __logf(s);
    float* orow = out + (size_t)row * 64;
    orow[lane]      = v0 - m - ls;
    orow[lane + 32] = v1 - m - ls;
}

// ---------------------------------------------------------------------------

static inline size_t align256(size_t x) { return (x + 255) & ~(size_t)255; }
static inline int cdiv(long long a, long long b) { return (int)((a + b - 1) / b); }

extern "C" void kernel_launch(void* const* d_in, const int* in_sizes, int n_in,
                              void* d_out, int out_size, void* d_ws, size_t ws_size,
                              hipStream_t stream) {
    // Inputs (setup_inputs order): x, edge_index, edge_weight, W1,b1, W2,b2, W3,b3, Wl,bl
    const float* x  = (const float*)d_in[0];
    const int*   ei = (const int*)  d_in[1];
    const float* ew = (const float*)d_in[2];
    const float* W[3]  = { (const float*)d_in[3], (const float*)d_in[5], (const float*)d_in[7] };
    const float* bb[3] = { (const float*)d_in[4], (const float*)d_in[6], (const float*)d_in[8] };
    const float* Wl = (const float*)d_in[9];
    const float* bl = (const float*)d_in[10];

    const int F    = 128;                      // Din == H == 128
    const int N    = in_sizes[0] / F;          // 50000
    const int E    = in_sizes[1] / 2;          // 500000
    const int Dout = out_size / N;             // 64
    const int* src = ei;
    const int* dst = ei + E;

    // Workspace partition
    char*   p    = (char*)d_ws;
    float*  deg  = (float*)p;  p += align256((size_t)N * 4);
    float*  dinv = (float*)p;  p += align256((size_t)N * 4);
    bf16_t* Xb   = (bf16_t*)p; p += align256((size_t)N * F * 2);
    float*  Hbuf = (float*)p;  p += align256((size_t)N * F * 4);
    float*  AGG  = (float*)p;  p += align256((size_t)N * F * 4);
    bf16_t* Wb[3];
    for (int l = 0; l < 3; ++l) { Wb[l] = (bf16_t*)p; p += align256((size_t)F * F * 2); }
    bf16_t* Wlb  = (bf16_t*)p;  p += align256((size_t)F * Dout * 2);
    (void)ws_size; (void)n_in;

    // Normalization: deg = 1 + sum(ew at dst); dinv = deg^-1/2
    k_init_deg   <<<cdiv(N, THREADS), THREADS, 0, stream>>>(deg, N);
    k_deg_scatter<<<cdiv(E, THREADS), THREADS, 0, stream>>>(dst, ew, deg, E);
    k_dinv       <<<cdiv(N, THREADS), THREADS, 0, stream>>>(deg, dinv, N);

    // Weights: convert + pack into WMMA B-fragment layout (once)
    for (int l = 0; l < 3; ++l)
        k_pack_w<<<cdiv((long long)(F >> 4) * (F >> 5) * 32, THREADS), THREADS, 0, stream>>>(W[l], Wb[l], F, F);
    k_pack_w<<<cdiv((long long)(Dout >> 4) * (F >> 5) * 32, THREADS), THREADS, 0, stream>>>(Wl, Wlb, F, Dout);

    // Layer-1 input: x -> bf16 (no ReLU)
    k_cvt_bf16<<<cdiv((long long)N * F, THREADS), THREADS, 0, stream>>>(x, Xb, (size_t)N * F, 0);

    // Three GCN layers: GEMM -> (bias + self-loop) init -> edge scatter -> ReLU->bf16
    const int gemmBlocks = cdiv(N >> 4, 8);    // one wave per 16-row stripe
    for (int l = 0; l < 3; ++l) {
        k_gemm_wmma_bf16<128, 8><<<gemmBlocks, THREADS, 0, stream>>>(Xb, Wb[l], Hbuf, N);
        k_agg_init    <<<cdiv((long long)N * F, THREADS), THREADS, 0, stream>>>(Hbuf, dinv, bb[l], AGG, N, F);
        k_edge_scatter<<<cdiv(E, 8), THREADS, 0, stream>>>(Hbuf, src, dst, ew, dinv, AGG, E, F);
        k_cvt_bf16    <<<cdiv((long long)N * F, THREADS), THREADS, 0, stream>>>(AGG, Xb, (size_t)N * F, 1);
    }

    // Final linear (N x 64) + log_softmax
    k_gemm_wmma_bf16<128, 4><<<gemmBlocks, THREADS, 0, stream>>>(Xb, Wlb, Hbuf, N);
    k_logsoftmax64<<<cdiv(N, 8), THREADS, 0, stream>>>(Hbuf, bl, (float*)d_out, N);
}
